// ConvolutionalGraphNetwork_10720238371126
// MI455X (gfx1250) — matrix operand order repaired and checked by reading, not compiled, not run on hardware
//
#include <hip/hip_runtime.h>

typedef __attribute__((ext_vector_type(16))) __bf16 v16bf;
typedef __attribute__((ext_vector_type(8)))  __bf16 v8bf;
typedef __attribute__((ext_vector_type(8)))  float  v8f;

#define NROWS 16384

// ---------------------------------------------------------------------------
// Small GEMM: tT = (h @ W)^T, output bf16, transposed so the adj-GEMM's
// B-fragments (two 8-wide K runs per lane) are contiguous loads.
// Cost is negligible (<0.6 GFLOP total across layers).
// ---------------------------------------------------------------------------
template<int K, int NCOL>
__global__ __launch_bounds__(256) void feat_gemm(
    const float* __restrict__ h, const float* __restrict__ W,
    __bf16* __restrict__ tT)
{
  __shared__ float Ws[K * NCOL];
  const int tid = threadIdx.x;
  for (int i = tid; i < K * NCOL; i += 256) Ws[i] = W[i];
  __syncthreads();

  const long row = (long)blockIdx.x * 256 + tid;
  float acc[NCOL];
#pragma unroll
  for (int j = 0; j < NCOL; ++j) acc[j] = 0.0f;

  const float* hr = h + row * K;
  for (int k = 0; k < K; ++k) {
    const float hk = hr[k];
#pragma unroll
    for (int j = 0; j < NCOL; ++j)
      acc[j] = fmaf(hk, Ws[k * NCOL + j], acc[j]);
  }
#pragma unroll
  for (int j = 0; j < NCOL; ++j)
    tT[(long)j * NROWS + row] = (__bf16)acc[j];
}

// ---------------------------------------------------------------------------
// Big GEMM: out = adj @ t + bias.  adj streamed fp32 from HBM (1 GiB/pass),
// converted to bf16 in-register, double-buffered through LDS, consumed by
// v_wmma_f32_16x16x32_bf16.  AI = 32 FLOP/B -> memory-bound at 23.3 TB/s.
// B tile (already bf16) is staged with GLOBAL_LOAD_ASYNC_TO_LDS_B128
// (ASYNCcnt), issued one tile ahead and fenced with s_wait_asynccnt before
// the single per-tile barrier.
// ---------------------------------------------------------------------------
template<int NCOL>
__global__ __launch_bounds__(128) void adj_gemm(
    const float* __restrict__ adj,      // NROWS x NROWS fp32
    const __bf16* __restrict__ tT,      // NCOL x NROWS bf16 (transposed t)
    const float* __restrict__ bias,     // NCOL
    float* __restrict__ out)            // NROWS x NCOL fp32
{
  constexpr int BR  = 64;          // rows per block
  constexpr int KT  = 64;          // K tile (two 32-wide WMMA K-steps)
  constexpr int LDA = 72;          // LDS row stride, bf16 elems (144 B:
                                   // 16B-aligned rows; 36-dword stride ->
                                   // 16 distinct bank residues per half-wave)
  constexpr int NB  = NCOL / 16;   // accumulator blocks per wave (4 or 1)

  __shared__ __bf16 As[2][BR * LDA];
  __shared__ __bf16 Bs[2][NCOL * LDA];

  const int tid  = threadIdx.x;
  const int wave = tid >> 5;
  const int lane = tid & 31;
  const int lr   = lane & 15;      // lane-in-halfwave
  const int kh   = lane >> 4;      // half-wave select

  const long rowBase = (long)blockIdx.x * BR;

  // --- A staging: 64x64 fp32 tile, 32 floats (8 float4) per thread ---
  const int aRow = tid >> 1;             // 0..63
  const int aCol = (tid & 1) * 32;       // 0 or 32
  const float* gA = adj + (rowBase + aRow) * (long)NROWS + aCol;

  float4 ar[8];                          // staged adj fp32

  auto loadRegsA = [&](int k0) {
#pragma unroll
    for (int i = 0; i < 8; ++i)
      ar[i] = *(const float4*)(gA + k0 + 4 * i);
  };

  auto stageA = [&](int buf) {
#pragma unroll
    for (int g = 0; g < 4; ++g) {        // 2 float4 -> 1 v8bf (packed cvt)
      const float4 f0 = ar[2 * g], f1 = ar[2 * g + 1];
      v8bf v{};
      v[0] = (__bf16)f0.x; v[1] = (__bf16)f0.y;
      v[2] = (__bf16)f0.z; v[3] = (__bf16)f0.w;
      v[4] = (__bf16)f1.x; v[5] = (__bf16)f1.y;
      v[6] = (__bf16)f1.z; v[7] = (__bf16)f1.w;
      *(v8bf*)(&As[buf][aRow * LDA + aCol + 8 * g]) = v;   // ds_store_b128
    }
  };

  // --- B staging: async global->LDS, NB ops of 128 lanes x 16 B.
  //     Op i covers rows 16i..16i+15 of the NCOL x 64 bf16 tile. ---
  const int bRow16 = tid >> 3;           // 0..15 row within 16-row group
  const int bCol   = (tid & 7) * 8;      // 8 bf16 = 16 B per lane

  auto asyncLoadB = [&](int k0, int buf) {
#pragma unroll
    for (int i = 0; i < NB; ++i) {
      const int row = i * 16 + bRow16;
      const __bf16* g = tT + (long)row * NROWS + k0 + bCol;
      const unsigned ldsOff =
          (unsigned)(size_t)(&Bs[buf][row * LDA + bCol]);  // LDS byte offset
      asm volatile("global_load_async_to_lds_b128 %0, %1, off"
                   :: "v"(ldsOff), "v"(g) : "memory");
    }
  };

  auto waitAsync = [&]() {
#if __has_builtin(__builtin_amdgcn_s_wait_asynccnt)
    __builtin_amdgcn_s_wait_asynccnt(0);
#else
    asm volatile("s_wait_asynccnt 0x0" ::: "memory");
#endif
  };

  // prologue: tile 0 -> LDS buf 0 (B async, A via regs), tile 1 A -> regs
  loadRegsA(0);
  asyncLoadB(0, 0);
  stageA(0);
  loadRegsA(KT);
  waitAsync();
  __syncthreads();

  v8f acc[NB];
#pragma unroll
  for (int j = 0; j < NB; ++j) acc[j] = {};

  const int arw = wave * 16 + lr;

  for (int k0 = 0; k0 < NROWS; k0 += KT) {
    const int p = (k0 / KT) & 1;
    const bool more = (k0 + KT) < NROWS;

    // issue next B tile early: async copy overlaps the whole compute phase
    if (more) asyncLoadB(k0 + KT, 1 - p);

    // ---- compute: gather fragments, then WMMAs ----
#pragma unroll
    for (int s = 0; s < 2; ++s) {        // two K=32 sub-steps
      // A fragment: lane<16 holds K {0..7,16..23}, lane>=16 K {8..15,24..31}
      const v8bf a0 = *(const v8bf*)(&As[p][arw * LDA + 32 * s + 8 * kh]);
      const v8bf a1 = *(const v8bf*)(&As[p][arw * LDA + 32 * s + 16 + 8 * kh]);
      const v16bf a = __builtin_shufflevector(a0, a1,
          0,1,2,3,4,5,6,7,8,9,10,11,12,13,14,15);

      v16bf bb[NB];
#pragma unroll
      for (int j = 0; j < NB; ++j) {
        // B fragment: lane<16 -> col lr, K {0..7,16..23}; lane>=16 K {8..15,24..31}
        const int n = j * 16 + lr;
        const v8bf b0 = *(const v8bf*)(&Bs[p][n * LDA + 32 * s + 8 * kh]);
        const v8bf b1 = *(const v8bf*)(&Bs[p][n * LDA + 32 * s + 16 + 8 * kh]);
        bb[j] = __builtin_shufflevector(b0, b1,
            0,1,2,3,4,5,6,7,8,9,10,11,12,13,14,15);
      }
#pragma unroll
      for (int j = 0; j < NB; ++j)
        acc[j] = __builtin_amdgcn_wmma_f32_16x16x32_bf16(
            false, a, false, bb[j], (short)0, acc[j], false, false);
    }

    // ---- stage A tile k0+KT into the other buffer, fetch tile k0+2*KT ----
    if (more) {
      stageA(1 - p);
      if (k0 + 2 * KT < NROWS) {
        loadRegsA(k0 + 2 * KT);
        __builtin_prefetch(gA + k0 + 3 * KT, 0, 0);  // global_prefetch_b8
      }
    }
    waitAsync();                         // B tile k0+KT resident in LDS
    __syncthreads();                     // single barrier per K-tile
  }

  // epilogue: C/D layout is VGPR v -> M = v + 8*kh, N = lr
#pragma unroll
  for (int j = 0; j < NB; ++j) {
    const int  col  = j * 16 + lr;
    const float bv  = bias[col];
    const long orow = rowBase + wave * 16 + 8 * kh;
#pragma unroll
    for (int v = 0; v < 8; ++v)
      out[(orow + v) * NCOL + col] = acc[j][v] + bv;
  }
}

// ---------------------------------------------------------------------------
extern "C" void kernel_launch(void* const* d_in, const int* in_sizes, int n_in,
                              void* d_out, int out_size, void* d_ws, size_t ws_size,
                              hipStream_t stream)
{
  const float* x     = (const float*)d_in[0];
  const float* adj   = (const float*)d_in[1];
  const float* W_in  = (const float*)d_in[2];
  const float* b_in  = (const float*)d_in[3];
  const float* W_h   = (const float*)d_in[4];   // [2,64,64]
  const float* b_h   = (const float*)d_in[5];   // [2,64]
  const float* W_out = (const float*)d_in[6];
  const float* b_out = (const float*)d_in[7];
  float* out = (float*)d_out;

  char* ws = (char*)d_ws;
  __bf16* tT = (__bf16*)ws;               // 64*16384*2B = 2 MB
  float*  h1 = (float*)(ws + (4u << 20)); // 16384*64*4B = 4 MB
  float*  h2 = (float*)(ws + (8u << 20)); // 4 MB

  const dim3 fb(256), fg(NROWS / 256);    // feat_gemm
  const dim3 ab(128), ag(NROWS / 64);     // adj_gemm

  // layer 0: h1 = adj @ (x @ W_in) + b_in
  feat_gemm<128, 64><<<fg, fb, 0, stream>>>(x, W_in, tT);
  adj_gemm<64><<<ag, ab, 0, stream>>>(adj, tT, b_in, h1);
  // hidden layer 0
  feat_gemm<64, 64><<<fg, fb, 0, stream>>>(h1, W_h, tT);
  adj_gemm<64><<<ag, ab, 0, stream>>>(adj, tT, b_h, h2);
  // hidden layer 1
  feat_gemm<64, 64><<<fg, fb, 0, stream>>>(h2, W_h + 64 * 64, tT);
  adj_gemm<64><<<ag, ab, 0, stream>>>(adj, tT, b_h + 64, h1);
  // output layer: out = adj @ (h1 @ W_out) + b_out
  feat_gemm<64, 16><<<fg, fb, 0, stream>>>(h1, W_out, tT);
  adj_gemm<16><<<ag, ab, 0, stream>>>(adj, tT, b_out, out);
}
